// ScaledDotMultiHeadAttention_22454089023607
// MI455X (gfx1250) — matrix operand order repaired and checked by reading, not compiled, not run on hardware
//
#include <hip/hip_runtime.h>
#include <hip/hip_bf16.h>
#include <cstdint>
#include <cstddef>

typedef __bf16 bf16;
typedef __attribute__((ext_vector_type(4)))  bf16  v4bf;
typedef __attribute__((ext_vector_type(8)))  bf16  v8bf;
typedef __attribute__((ext_vector_type(16))) bf16  v16bf;
typedef __attribute__((ext_vector_type(8)))  float v8f;

typedef unsigned int u32;
typedef u32 __attribute__((ext_vector_type(4))) u32x4;
typedef int __attribute__((ext_vector_type(8))) i32x8;
typedef int __attribute__((ext_vector_type(4))) i32x4;

static constexpr int BB = 4, TT = 2048, DD = 1024, HH = 16, HD = 64;
static constexpr int MTOT = BB * TT;   // 8192 rows of x

#if __has_builtin(__builtin_amdgcn_tensor_load_to_lds)
#define USE_TDM 1
#endif

// ---------------------------------------------------------------------------
// WMMA helpers (CDNA5, wave32): D(16x16,f32) = A(16x32,bf16) x B(32x16,bf16) + C
// ---------------------------------------------------------------------------
__device__ __forceinline__ v8f wmma_bf16(v16bf a, v16bf b, v8f c) {
  // 8 args: (neg_a, A, neg_b, B, c_mod, C, reuse_a, reuse_b)
  return __builtin_amdgcn_wmma_f32_16x16x32_bf16(false, a, false, b, (short)0, c,
                                                 false, false);
}

// A-fragment: 16x32 (MxK), source row-major with leading dim ld.
// ISA layout: lanes 0-15 = rows 0-15 holding K {0..7, 16..23};
//             lanes 16-31 = rows 0-15 holding K {8..15, 24..31}.
__device__ __forceinline__ v16bf load_frag_a(const bf16* p, int ld, int lane) {
  const bf16* q = p + (size_t)(lane & 15) * ld + ((lane >> 4) << 3);
  v8bf lo = *(const v8bf*)q;
  v8bf hi = *(const v8bf*)(q + 16);
  v16bf r;
#pragma unroll
  for (int i = 0; i < 8; ++i) { r[i] = lo[i]; r[i + 8] = hi[i]; }
  return r;
}

// B-fragment: 32x16 (KxN), sourced from Bt = B^T stored row-major (N x K, ld).
// ISA layout: lanes 0-15 = col n holding K {0..7, 16..23};
//             lanes 16-31 = col n holding K {8..15, 24..31}.
__device__ __forceinline__ v16bf load_frag_b(const bf16* p, int ld, int lane) {
  const bf16* q = p + (size_t)(lane & 15) * ld + ((lane >> 4) << 3);
  v8bf lo = *(const v8bf*)q;
  v8bf hi = *(const v8bf*)(q + 16);
  v16bf r;
#pragma unroll
  for (int i = 0; i < 8; ++i) { r[i] = lo[i]; r[i + 8] = hi[i]; }
  return r;
}

// ---------------------------------------------------------------------------
// f32 -> bf16 conversion (vectorized)
// ---------------------------------------------------------------------------
__global__ __launch_bounds__(256) void f32_to_bf16_kernel(
    const float* __restrict__ src, bf16* __restrict__ dst, int n4) {
  int i = blockIdx.x * blockDim.x + threadIdx.x;
  if (i < n4) {
    float4 v = ((const float4*)src)[i];
    v4bf o;
    o[0] = (bf16)v.x; o[1] = (bf16)v.y; o[2] = (bf16)v.z; o[3] = (bf16)v.w;
    ((v4bf*)dst)[i] = o;
  }
}

// ---------------------------------------------------------------------------
// GEMM: C[m,n] = (sum_k A[m,k]*Bt[n,k] + bias[n]) * scale
// A: M x K row-major bf16.  Bt: N x K row-major bf16 (i.e. weights (N,K)).
// Grid: blockIdx.y = 64-column N panel, blockIdx.x * 8 waves = 32-row M tiles.
// The 64x32 B panel (4 KB) is staged into LDS, double-buffered:
//   - TDM path: one TENSOR_LOAD_TO_LDS (2D tile descriptor) issued by wave 0
//     per K-step, tracked with TENSORcnt (s_wait_tensorcnt).
//   - Fallback: per-thread GLOBAL_LOAD_ASYNC_TO_LDS_B128 (ASYNCcnt).
// The next panel's DMA overlaps the current panel's WMMAs.
// MODE 0: dst bf16 laid out (b, h, t, 64)        (Q / K for attention)
// MODE 1: dst bf16 laid out (b, h, 64, t)        (V transposed)
// MODE 2: dst f32 plain row-major M x N          (final output)
// ---------------------------------------------------------------------------
template <int MODE>
__global__ __launch_bounds__(256) void gemm_bf16_wmma(
    const bf16* __restrict__ A, const bf16* __restrict__ Bt,
    const float* __restrict__ bias, void* __restrict__ dstv,
    int M, int N, int K, float scale) {
  __shared__ __align__(16) bf16 ldsB[2][64 * 32];  // 2 x 4 KB B panels

  const int lane = threadIdx.x & 31;
  const int wloc = threadIdx.x >> 5;
  const int n0 = blockIdx.y << 6;
  const int m0 = (blockIdx.x * 8 + wloc) << 5;

  const u32 ldsBase = (u32)(uintptr_t)(&ldsB[0][0]);

#ifdef USE_TDM
  // One 2D-tile TDM descriptor per K-step: tile 32(k) x 64(n), elem = 2 B.
  auto issue_tdm = [&](int kk, int buf) {
    const uint64_t gaddr =
        (uint64_t)(uintptr_t)Bt + (((uint64_t)n0 * (u32)K + (u32)kk) << 1);
    u32x4 g0;
    g0[0] = 1u;                                   // count=1, user mode
    g0[1] = ldsBase + (u32)buf * (64u * 32u * 2u);// lds_addr (bytes)
    g0[2] = (u32)gaddr;                           // global tile addr [31:0]
    g0[3] = (u32)(gaddr >> 32) | (2u << 30);      // addr [56:32] | type=2
    i32x8 g1;
    g1[0] = (int)(1u << 16);                      // data_size = 1 -> 2 bytes
    g1[1] = (int)((u32)K << 16);                  // tensor_dim0[15:0]
    g1[2] = (int)(((u32)K >> 16) | ((u32)N << 16)); // dim0 hi | tensor_dim1 lo
    g1[3] = (int)(((u32)N >> 16) | (32u << 16));  // dim1 hi | tile_dim0 = 32
    g1[4] = 64;                                   // tile_dim1 = 64, tile_dim2=0
    g1[5] = K;                                    // tensor_dim0_stride[31:0]
    g1[6] = 0;                                    // stride hi | dim1_stride lo
    g1[7] = 0;
    i32x4 z4 = {0, 0, 0, 0};
    i32x8 z8 = {0, 0, 0, 0, 0, 0, 0, 0};
    __builtin_amdgcn_tensor_load_to_lds(g0, g1, z4, z4, z8, 0);
  };
#else
  // Per-thread async copy: thread t moves 16 B of the B panel.
  const int brow = threadIdx.x >> 2;              // 0..63
  const int bchk = threadIdx.x & 3;               // 0..3 (k chunk of 8)
  const u32 ldsOff = ldsBase + (u32)(brow * 32 + bchk * 8) * 2u;
  const u32 gBase = (u32)((n0 + brow) * K + bchk * 8) * 2u;
  auto issue_async = [&](int kk, int buf) {
    const u32 dst = ldsOff + (u32)buf * (64u * 32u * 2u);
    const u32 goff = gBase + ((u32)kk << 1);
    asm volatile("global_load_async_to_lds_b128 %0, %1, %2 offset:0"
                 :
                 : "v"(dst), "v"(goff), "s"(Bt)
                 : "memory");
  };
#endif

  v8f acc[2][4];
#pragma unroll
  for (int i = 0; i < 2; ++i)
#pragma unroll
    for (int j = 0; j < 4; ++j)
#pragma unroll
      for (int r = 0; r < 8; ++r) acc[i][j][r] = 0.0f;

  // Prologue: start DMA of the first panel.
#ifdef USE_TDM
  if (wloc == 0) issue_tdm(0, 0);
#else
  issue_async(0, 0);
#endif

  for (int kk = 0; kk < K; kk += 32) {
    const int cur = (kk >> 5) & 1;
    // Wait for the current panel's DMA, make it visible block-wide.
#ifdef USE_TDM
    if (wloc == 0) __builtin_amdgcn_s_wait_tensorcnt(0);
#else
    asm volatile("s_wait_asynccnt 0" ::: "memory");
#endif
    __syncthreads();
    // Kick off the next panel while we compute on this one.
    if (kk + 32 < K) {
#ifdef USE_TDM
      if (wloc == 0) issue_tdm(kk + 32, cur ^ 1);
#else
      issue_async(kk + 32, cur ^ 1);
#endif
    }

    v16bf a0 = load_frag_a(A + (size_t)m0 * K + kk, K, lane);
    v16bf a1 = load_frag_a(A + (size_t)(m0 + 16) * K + kk, K, lane);
#pragma unroll
    for (int j = 0; j < 4; ++j) {
      v16bf b = load_frag_b(&ldsB[cur][16 * j * 32], 32, lane);
      acc[0][j] = wmma_bf16(a0, b, acc[0][j]);
      acc[1][j] = wmma_bf16(a1, b, acc[1][j]);
    }
    __syncthreads();  // done reading ldsB[cur] before it is overwritten
  }

  const int col = lane & 15;
  const int rsel = (lane >> 4) << 3;   // +8 rows for upper half-wave
#pragma unroll
  for (int i = 0; i < 2; ++i) {
#pragma unroll
    for (int j = 0; j < 4; ++j) {
      const int n = n0 + 16 * j + col;
      const float bv = bias[n];
#pragma unroll
      for (int r = 0; r < 8; ++r) {
        const int m = m0 + 16 * i + rsel + r;
        const float val = (acc[i][j][r] + bv) * scale;
        if (MODE == 2) {
          ((float*)dstv)[(size_t)m * N + n] = val;
        } else {
          const int b = m / TT, t = m % TT;
          const int h = n >> 6, d = n & 63;
          const bf16 o = (bf16)val;
          if (MODE == 0)
            ((bf16*)dstv)[(((size_t)b * HH + h) * TT + t) * HD + d] = o;
          else
            ((bf16*)dstv)[(((size_t)b * HH + h) * HD + d) * TT + t] = o;
        }
      }
    }
  }
}

// ---------------------------------------------------------------------------
// Flash attention (causal). One wave per 16-query tile of one (b,h).
// Q, K: (b,h,t,64) bf16 (Q pre-scaled by 1/sqrt(64)). Vt: (b,h,64,t) bf16.
// Out: (b*T, H*64) bf16 row-major.
// ---------------------------------------------------------------------------
__global__ __launch_bounds__(256) void attn_fwd_kernel(
    const bf16* __restrict__ Q, const bf16* __restrict__ Kc,
    const bf16* __restrict__ Vt, bf16* __restrict__ Out) {
  __shared__ __align__(16) bf16 ldsP[8][16 * 32];  // per-wave P staging (1 KB)

  const int lane = threadIdx.x & 31;
  const int wloc = threadIdx.x >> 5;
  const int wid  = blockIdx.x * 8 + wloc;
  const int qt = wid & (TT / 16 - 1);          // 128 q-tiles per head
  const int h  = (wid >> 7) & (HH - 1);
  const int b  = wid >> 11;
  const int q0 = qt << 4;

  const bf16* Qp = Q  + (((size_t)b * HH + h) * TT + q0) * HD;
  const bf16* Kp = Kc + ((size_t)b * HH + h) * TT * HD;
  const bf16* Vp = Vt + ((size_t)b * HH + h) * HD * TT;

  const v16bf qa0 = load_frag_a(Qp, HD, lane);        // d = 0..31
  const v16bf qa1 = load_frag_a(Qp + 32, HD, lane);   // d = 32..63

  float mrow[8], lrow[8];
  v8f o[4];
#pragma unroll
  for (int r = 0; r < 8; ++r) { mrow[r] = -1e30f; lrow[r] = 0.0f; }
#pragma unroll
  for (int dt = 0; dt < 4; ++dt)
#pragma unroll
    for (int r = 0; r < 8; ++r) o[dt][r] = 0.0f;

  const int col  = lane & 15;
  const int rsel = (lane >> 4) << 3;
  bf16* lp = ldsP[wloc];
  const int nblk = (q0 + 47) >> 5;   // causal: keys 0 .. q0+15

  for (int jb = 0; jb < nblk; ++jb) {
    const int j = jb << 5;

    // Prefetch next key block (global_prefetch_b8), one cacheline per lane
    if (jb + 1 < nblk)
      __builtin_prefetch(Kp + (size_t)(j + 32 + lane) * HD, 0, 3);

    // S tile: 16 queries x 32 keys (two 16x16 wmma column tiles, K over d)
    v8f s[2];
#pragma unroll
    for (int t2 = 0; t2 < 2; ++t2) {
      v16bf kb0 = load_frag_b(Kp + (size_t)(j + 16 * t2) * HD, HD, lane);
      v16bf kb1 = load_frag_b(Kp + (size_t)(j + 16 * t2) * HD + 32, HD, lane);
      v8f z;
#pragma unroll
      for (int r = 0; r < 8; ++r) z[r] = 0.0f;
      s[t2] = wmma_bf16(qa0, kb0, z);
      s[t2] = wmma_bf16(qa1, kb1, s[t2]);
    }

    // Causal mask (only diagonal blocks need it)
    if (j + 31 > q0) {
#pragma unroll
      for (int t2 = 0; t2 < 2; ++t2)
#pragma unroll
        for (int r = 0; r < 8; ++r) {
          const int qrow = q0 + rsel + r;
          const int key  = j + 16 * t2 + col;
          if (key > qrow) s[t2][r] = -1e30f;
        }
    }

    // Online softmax update (row stats replicated across 16-lane groups)
#pragma unroll
    for (int r = 0; r < 8; ++r) {
      float bmax = fmaxf(s[0][r], s[1][r]);
      bmax = fmaxf(bmax, __shfl_xor(bmax, 1, 32));
      bmax = fmaxf(bmax, __shfl_xor(bmax, 2, 32));
      bmax = fmaxf(bmax, __shfl_xor(bmax, 4, 32));
      bmax = fmaxf(bmax, __shfl_xor(bmax, 8, 32));
      const float newm = fmaxf(mrow[r], bmax);
      const float resc = __expf(mrow[r] - newm);
      const float p0 = __expf(s[0][r] - newm);
      const float p1 = __expf(s[1][r] - newm);
      s[0][r] = p0; s[1][r] = p1;
      float ls = p0 + p1;
      ls += __shfl_xor(ls, 1, 32);
      ls += __shfl_xor(ls, 2, 32);
      ls += __shfl_xor(ls, 4, 32);
      ls += __shfl_xor(ls, 8, 32);
      lrow[r] = lrow[r] * resc + ls;
      mrow[r] = newm;
#pragma unroll
      for (int dt = 0; dt < 4; ++dt) o[dt][r] *= resc;
    }

    // Stage P (16x32 bf16) through LDS to re-fragment as an A matrix
#pragma unroll
    for (int t2 = 0; t2 < 2; ++t2)
#pragma unroll
      for (int r = 0; r < 8; ++r)
        lp[(rsel + r) * 32 + 16 * t2 + col] = (bf16)s[t2][r];

    const v16bf pa = load_frag_a(lp, 32, lane);

    // O += P(16x32) @ V(32x64): 4 wmma, B-fragments from V^T rows (d, t)
#pragma unroll
    for (int dt = 0; dt < 4; ++dt) {
      v16bf vb = load_frag_b(Vp + (size_t)(16 * dt) * TT + j, TT, lane);
      o[dt] = wmma_bf16(pa, vb, o[dt]);
    }
  }

  // Normalize and write out (b*T + t, h*64 + d) bf16
#pragma unroll
  for (int r = 0; r < 8; ++r) {
    const float inv = 1.0f / lrow[r];
    const int t = q0 + rsel + r;
#pragma unroll
    for (int dt = 0; dt < 4; ++dt) {
      const size_t idx = ((size_t)b * TT + t) * DD + h * HD + dt * 16 + col;
      Out[idx] = (bf16)(o[dt][r] * inv);
    }
  }
}

// ---------------------------------------------------------------------------
// Host launcher
// ---------------------------------------------------------------------------
extern "C" void kernel_launch(void* const* d_in, const int* in_sizes, int n_in,
                              void* d_out, int out_size, void* d_ws, size_t ws_size,
                              hipStream_t stream) {
  const float* x  = (const float*)d_in[0];
  // d_in[1] = causal mask (ignored; causality handled analytically)
  const float* Wq = (const float*)d_in[2];
  const float* bq = (const float*)d_in[3];
  const float* Wk = (const float*)d_in[4];
  const float* bk = (const float*)d_in[5];
  const float* Wv = (const float*)d_in[6];
  const float* bv = (const float*)d_in[7];
  const float* Wo = (const float*)d_in[8];
  const float* bo = (const float*)d_in[9];

  char* ws = (char*)d_ws;
  size_t off = 0;
  auto carve = [&](size_t bytes) -> void* {
    void* p = ws + off;
    off += (bytes + 255) & ~(size_t)255;
    return p;
  };
  bf16* xb   = (bf16*)carve((size_t)MTOT * DD * sizeof(bf16));  // 16 MB
  bf16* wqb  = (bf16*)carve((size_t)DD * DD * sizeof(bf16));    //  2 MB
  bf16* wkb  = (bf16*)carve((size_t)DD * DD * sizeof(bf16));
  bf16* wvb  = (bf16*)carve((size_t)DD * DD * sizeof(bf16));
  bf16* wob  = (bf16*)carve((size_t)DD * DD * sizeof(bf16));
  bf16* Qb   = (bf16*)carve((size_t)MTOT * DD * sizeof(bf16));  // (b,h,t,64)
  bf16* Kb   = (bf16*)carve((size_t)MTOT * DD * sizeof(bf16));  // (b,h,t,64)
  bf16* Vtb  = (bf16*)carve((size_t)MTOT * DD * sizeof(bf16));  // (b,h,64,t)
  bf16* attn = (bf16*)carve((size_t)MTOT * DD * sizeof(bf16));  // (b*t, h*64)

  auto cvt = [&](const float* s, bf16* d, int n) {
    f32_to_bf16_kernel<<<(n / 4 + 255) / 256, 256, 0, stream>>>(s, d, n / 4);
  };
  cvt(x, xb, MTOT * DD);
  cvt(Wq, wqb, DD * DD);
  cvt(Wk, wkb, DD * DD);
  cvt(Wv, wvb, DD * DD);
  cvt(Wo, wob, DD * DD);

  // Grid: x = 8192 rows / (8 waves * 32 rows) = 32, y = 1024 / 64 = 16
  dim3 ggrid(MTOT / 256, DD / 64);
  gemm_bf16_wmma<0><<<ggrid, 256, 0, stream>>>(xb, wqb, bq, (void*)Qb,
                                               MTOT, DD, DD, 0.125f);
  gemm_bf16_wmma<0><<<ggrid, 256, 0, stream>>>(xb, wkb, bk, (void*)Kb,
                                               MTOT, DD, DD, 1.0f);
  gemm_bf16_wmma<1><<<ggrid, 256, 0, stream>>>(xb, wvb, bv, (void*)Vtb,
                                               MTOT, DD, DD, 1.0f);

  // 8192 q-tiles -> 1024 blocks of 8 waves
  attn_fwd_kernel<<<(BB * HH * (TT / 16)) / 8, 256, 0, stream>>>(Qb, Kb, Vtb, attn);

  gemm_bf16_wmma<2><<<ggrid, 256, 0, stream>>>(attn, wob, bo, d_out,
                                               MTOT, DD, DD, 1.0f);
}
